// Decoder_58755152609631
// MI455X (gfx1250) — matrix operand order repaired and checked by reading, not compiled, not run on hardware
//
#include <hip/hip_runtime.h>

typedef __attribute__((ext_vector_type(16))) __bf16 v16bf;
typedef __attribute__((ext_vector_type(8)))  __bf16 v8bf;
typedef __attribute__((ext_vector_type(4)))  __bf16 v4bf;
typedef __attribute__((ext_vector_type(8)))  float  v8f;

#define BM 64
#define BN 128
#define KB 32
#define KST 40   // padded LDS k-stride (elems): 80B rows keep 16B-aligned b128 ds loads

// EPI 0: out[m*ldo+n] = acc + bias[n]
// EPI 1: attention score: atomicAdd(out[m], sum_n tanh(acc + bias[n] + q[b][n]) * Vw[n])
template<int EPI>
__global__ __launch_bounds__(256)
void gemm_bf16(const float* __restrict__ A, int lda,
               const float* __restrict__ B, int ldb,
               const float* __restrict__ bias,
               float* __restrict__ out, int ldo,
               int K,
               const float* __restrict__ qbuf,
               const float* __restrict__ Vw)
{
    __shared__ __align__(16) __bf16 As[2][BM][KST];
    __shared__ __align__(16) __bf16 Bs[2][BN][KST];

    const int tid  = threadIdx.x;
    const int lane = tid & 31;
    const int wave = tid >> 5;      // 0..7
    const int wm   = wave & 3;      // 4 x 16 rows
    const int wn   = wave >> 2;     // 2 x 64 cols
    const int half = lane >> 4;
    const int l16  = lane & 15;

    const int m0 = blockIdx.y * BM;
    const int n0 = blockIdx.x * BN;

    v8f acc[4];
    #pragma unroll
    for (int t = 0; t < 4; ++t)
        #pragma unroll
        for (int r = 0; r < 8; ++r) acc[t][r] = 0.f;

    // A staging: 64 rows x 32 k, 8 consecutive k per thread -> one b128 store
    const int arow = tid >> 2;            // 0..63
    const int acol = (tid & 3) * 8;       // 0,8,16,24
    // B staging: 32 k x 128 n, 4k x 4n per thread -> packed-k b64 stores
    const int bk   = (tid >> 5) * 4;      // 0,4,...,28
    const int bn   = (tid & 31) * 4;      // 0..124

    float ar[8];
    float br[4][4];

    auto loadG = [&](int k0) {
        const float* asrc = A + (long)(m0 + arow) * lda + k0 + acol;
        float4 t0 = *(const float4*)asrc;
        float4 t1 = *(const float4*)(asrc + 4);
        ar[0] = t0.x; ar[1] = t0.y; ar[2] = t0.z; ar[3] = t0.w;
        ar[4] = t1.x; ar[5] = t1.y; ar[6] = t1.z; ar[7] = t1.w;
        const float* bsrc = B + (long)(k0 + bk) * ldb + n0 + bn;
        #pragma unroll
        for (int kk = 0; kk < 4; ++kk) {
            float4 t = *(const float4*)(bsrc + (long)kk * ldb);
            br[kk][0] = t.x; br[kk][1] = t.y; br[kk][2] = t.z; br[kk][3] = t.w;
        }
    };
    auto stage = [&](int p) {
        v8bf av;
        #pragma unroll
        for (int i = 0; i < 8; ++i) av[i] = (__bf16)ar[i];
        *(v8bf*)&As[p][arow][acol] = av;
        #pragma unroll
        for (int j = 0; j < 4; ++j) {
            v4bf bv;
            #pragma unroll
            for (int kk = 0; kk < 4; ++kk) bv[kk] = (__bf16)br[kk][j];
            *(v4bf*)&Bs[p][bn + j][bk] = bv;
        }
    };

    int p = 0;
    loadG(0);
    for (int k0 = 0; k0 < K; k0 += KB) {
        stage(p);
        __syncthreads();
        if (k0 + KB < K) loadG(k0 + KB);   // prefetch next tile while computing

        // A fragment: element i -> K = 16*(i>=8) + 8*half + (i&7)
        v16bf a;
        {
            const __bf16* ap = &As[p][wm * 16 + l16][0];
            v8bf lo = *(const v8bf*)(ap + 8 * half);
            v8bf hi = *(const v8bf*)(ap + 16 + 8 * half);
            #pragma unroll
            for (int i = 0; i < 8; ++i) { a[i] = lo[i]; a[i + 8] = hi[i]; }
        }
        // load ALL B fragments first (distinct regs) so one s_wait_dscnt
        // covers them and the 4 WMMAs can issue back-to-back
        v16bf bf[4];
        #pragma unroll
        for (int t = 0; t < 4; ++t) {
            const __bf16* bp = &Bs[p][wn * 64 + t * 16 + l16][0];
            v8bf lo = *(const v8bf*)(bp + 8 * half);
            v8bf hi = *(const v8bf*)(bp + 16 + 8 * half);
            #pragma unroll
            for (int i = 0; i < 8; ++i) { bf[t][i] = lo[i]; bf[t][i + 8] = hi[i]; }
        }
        #pragma unroll
        for (int t = 0; t < 4; ++t)
            acc[t] = __builtin_amdgcn_wmma_f32_16x16x32_bf16(
                false, a, false, bf[t], (short)0, acc[t], false, false);
        p ^= 1;
        // no second barrier: buffer reuse distance is 2 iterations, and the
        // next iteration's barrier orders all waves' compute before overwrite
    }

    if constexpr (EPI == 0) {
        #pragma unroll
        for (int t = 0; t < 4; ++t) {
            const int n = n0 + wn * 64 + t * 16 + l16;
            const float bv = bias[n];
            #pragma unroll
            for (int r = 0; r < 8; ++r) {
                const int m = m0 + wm * 16 + half * 8 + r;
                out[(long)m * ldo + n] = acc[t][r] + bv;
            }
        }
    } else {
        const int b = m0 >> 8;  // S == 256, BM divides S
        float part[8];
        #pragma unroll
        for (int r = 0; r < 8; ++r) part[r] = 0.f;
        #pragma unroll
        for (int t = 0; t < 4; ++t) {
            const int n = n0 + wn * 64 + t * 16 + l16;
            const float add = bias[n] + qbuf[b * 1024 + n];
            const float vw  = Vw[n];
            #pragma unroll
            for (int r = 0; r < 8; ++r)
                part[r] += tanhf(acc[t][r] + add) * vw;
        }
        #pragma unroll
        for (int msk = 1; msk < 16; msk <<= 1)
            #pragma unroll
            for (int r = 0; r < 8; ++r)
                part[r] += __shfl_xor(part[r], msk, 32);
        if (l16 == 0) {
            #pragma unroll
            for (int r = 0; r < 8; ++r)
                atomicAdd(&out[m0 + wm * 16 + half * 8 + r], part[r]);
        }
    }
}

__global__ __launch_bounds__(256)
void softmax_context(const float* __restrict__ score,
                     const float* __restrict__ enc,
                     float* __restrict__ ctx)
{
    __shared__ float sc[256];
    __shared__ float at[256];
    const int b = blockIdx.x, tid = threadIdx.x;
    sc[tid] = score[b * 256 + tid];
    __syncthreads();
    float mx = -1e30f;
    for (int s = 0; s < 256; ++s) mx = fmaxf(mx, sc[s]);
    at[tid] = __expf(sc[tid] - mx);
    __syncthreads();
    float sum = 0.f;
    for (int s = 0; s < 256; ++s) sum += at[s];
    const float inv = 1.f / sum;
    #pragma unroll
    for (int j = 0; j < 4; ++j) {
        const int u = tid + j * 256;
        float a0 = 0.f;
        for (int s = 0; s < 256; ++s)
            a0 += at[s] * enc[((long)b * 256 + s) * 1024 + u];
        ctx[b * 1024 + u] = a0 * inv;
    }
}

__global__ void concat_emb(const float* __restrict__ ctx,
                           const float* __restrict__ emb,
                           const int* __restrict__ x,
                           float* __restrict__ out_)
{
    const int b = blockIdx.x;
    const int row = x[b];
    for (int u = threadIdx.x; u < 1280; u += blockDim.x)
        out_[b * 1280 + u] = (u < 1024) ? ctx[b * 1024 + u]
                                        : emb[(long)row * 256 + (u - 1024)];
}

__global__ void concat_h(const float* __restrict__ ctx,
                         const float* __restrict__ h,
                         float* __restrict__ out_)
{
    const int b = blockIdx.x;
    for (int u = threadIdx.x; u < 2048; u += blockDim.x)
        out_[b * 2048 + u] = (u < 1024) ? ctx[b * 1024 + u]
                                        : h[b * 1024 + (u - 1024)];
}

__global__ void lstm_gates(const float* __restrict__ z, float* __restrict__ h)
{
    const int idx = blockIdx.x * blockDim.x + threadIdx.x;   // 131072
    const int b = idx >> 10, u = idx & 1023;
    const float* zr = z + (long)b * 4096;
    const float gi = zr[u], gg = zr[2048 + u], go = zr[3072 + u];
    const float si = 1.f / (1.f + __expf(-gi));
    const float so = 1.f / (1.f + __expf(-go));
    const float c  = si * tanhf(gg);
    h[idx] = so * tanhf(c);
}

__global__ void zero_f32(float* __restrict__ p, int n)
{
    const int i = blockIdx.x * blockDim.x + threadIdx.x;
    if (i < n) p[i] = 0.f;
}

extern "C" void kernel_launch(void* const* d_in, const int* in_sizes, int n_in,
                              void* d_out, int out_size, void* d_ws, size_t ws_size,
                              hipStream_t stream) {
    const int*   x      = (const int*)  d_in[0];
    const float* hidden = (const float*)d_in[1];
    const float* enc    = (const float*)d_in[2];
    const float* W1_w   = (const float*)d_in[3];
    const float* W1_b   = (const float*)d_in[4];
    const float* W2_w   = (const float*)d_in[5];
    const float* W2_b   = (const float*)d_in[6];
    const float* V_w    = (const float*)d_in[7];
    // d_in[8] = V_b: softmax is shift-invariant, score unused elsewhere -> skip
    const float* emb    = (const float*)d_in[9];
    const float* l1_W   = (const float*)d_in[10];
    const float* l1_b   = (const float*)d_in[11];
    const float* l2_W   = (const float*)d_in[12];
    const float* l2_b   = (const float*)d_in[13];
    const float* l3_W   = (const float*)d_in[14];
    const float* l3_b   = (const float*)d_in[15];
    const float* l4_W   = (const float*)d_in[16];
    const float* l4_b   = (const float*)d_in[17];
    const float* fc_w   = (const float*)d_in[18];
    const float* fc_b   = (const float*)d_in[19];

    float* ws    = (float*)d_ws;
    float* q     = ws;                    // 128*1024
    float* ctx   = ws + 131072;           // 128*1024
    float* score = ws + 262144;           // 128*256 = 32768
    float* inp   = ws + 294912;           // 128*2048 (l1 uses 128*1280)
    float* z     = ws + 557056;           // 128*4096

    float* logits = (float*)d_out;                 // 128*32000
    float* h      = (float*)d_out + 128 * 32000;   // 128*1024

    // 1. zero score accumulator
    zero_f32<<<128, 256, 0, stream>>>(score, 32768);

    // 2. q = hidden @ W1_w + W1_b          (128 x 1024 x 1024)
    gemm_bf16<0><<<dim3(8, 2), 256, 0, stream>>>(
        hidden, 1024, W1_w, 1024, W1_b, q, 1024, 1024, nullptr, nullptr);

    // 3. fused keys/score: score[b,s] += sum_n tanh(enc@W2 + b2 + q) * V_w
    //    (32768 x 1024 x 1024, the big GEMM; keys never materialized)
    gemm_bf16<1><<<dim3(8, 512), 256, 0, stream>>>(
        enc, 1024, W2_w, 1024, W2_b, score, 0, 1024, q, V_w);

    // 4. softmax over S + context reduction
    softmax_context<<<128, 256, 0, stream>>>(score, enc, ctx);

    // 5. LSTM layer 1: inp = [ctx | emb[x]] ; z = inp @ l1_W + b ; gates
    concat_emb<<<128, 256, 0, stream>>>(ctx, emb, x, inp);
    gemm_bf16<0><<<dim3(32, 2), 256, 0, stream>>>(
        inp, 1280, l1_W, 4096, l1_b, z, 4096, 1280, nullptr, nullptr);
    lstm_gates<<<512, 256, 0, stream>>>(z, h);

    // 6-8. LSTM layers 2..4: inp = [ctx | h]
    const float* lW[3] = { l2_W, l3_W, l4_W };
    const float* lb[3] = { l2_b, l3_b, l4_b };
    for (int i = 0; i < 3; ++i) {
        concat_h<<<128, 256, 0, stream>>>(ctx, h, inp);
        gemm_bf16<0><<<dim3(32, 2), 256, 0, stream>>>(
            inp, 2048, lW[i], 4096, lb[i], z, 4096, 2048, nullptr, nullptr);
        lstm_gates<<<512, 256, 0, stream>>>(z, h);
    }

    // 9. logits = h @ fc_w + fc_b          (128 x 32000 x 1024)
    gemm_bf16<0><<<dim3(250, 2), 256, 0, stream>>>(
        h, 1024, fc_w, 32000, fc_b, logits, 32000, 1024, nullptr, nullptr);
}
